// MambaBlock_32804960207127
// MI455X (gfx1250) — matrix operand order repaired and checked by reading, not compiled, not run on hardware
//
#include <hip/hip_runtime.h>
#include <hip/hip_bf16.h>

#define D_MODEL 1024
#define D_STATE 16
#define DT_RANK 64
#define D_INNER 2048
#define B_SZ    4
#define L_SEQ   4096
#define NTOK    (B_SZ * L_SEQ)   // 16384 tokens

typedef __attribute__((ext_vector_type(16))) __bf16        v16bf;
typedef __attribute__((ext_vector_type(8)))  float         v8f;
typedef __attribute__((ext_vector_type(4)))  unsigned int  v4u;

union FragBF {
    v16bf bf;
    v4u   q[2];   // 2 x 16B
};

__device__ __forceinline__ unsigned short f2bf(float f) {
    union { float f; unsigned u; } v; v.f = f;
    unsigned x = v.u;
    x += 0x7FFFu + ((x >> 16) & 1u);   // round-to-nearest-even
    return (unsigned short)(x >> 16);
}

// ---------------------------------------------------------------- converts
__global__ __launch_bounds__(256) void k_f32_to_bf16(const float* __restrict__ in,
                                                     unsigned short* __restrict__ out, int n) {
    int i = blockIdx.x * 256 + threadIdx.x;
    if (i < n) out[i] = f2bf(in[i]);
}

// slice columns [col0, col0+ncol) of a [nrow, ld_in] f32 matrix -> packed bf16 [nrow, ncol]
__global__ __launch_bounds__(256) void k_slice_bf16(const float* __restrict__ in, int ld_in, int col0,
                                                    unsigned short* __restrict__ out, int ncol, int nrow) {
    int i = blockIdx.x * 256 + threadIdx.x;
    if (i >= nrow * ncol) return;
    int r = i / ncol, c = i - r * ncol;
    out[i] = f2bf(in[(size_t)r * ld_in + col0 + c]);
}

// ---------------------------------------------------------------- layernorm -> bf16
__global__ __launch_bounds__(256) void k_layernorm(const float* __restrict__ x,
                                                   const float* __restrict__ gamma,
                                                   const float* __restrict__ beta,
                                                   unsigned short* __restrict__ xn) {
    const int row = blockIdx.x;
    const float* xr = x + (size_t)row * D_MODEL;
    float v[4], s = 0.f, sq = 0.f;
#pragma unroll
    for (int i = 0; i < 4; ++i) {
        int c = threadIdx.x + i * 256;
        v[i] = xr[c];
        s += v[i]; sq += v[i] * v[i];
    }
#pragma unroll
    for (int off = 16; off; off >>= 1) {
        s  += __shfl_xor(s,  off, 32);
        sq += __shfl_xor(sq, off, 32);
    }
    __shared__ float ss[8], ssq[8];
    if ((threadIdx.x & 31) == 0) { ss[threadIdx.x >> 5] = s; ssq[threadIdx.x >> 5] = sq; }
    __syncthreads();
    s = 0.f; sq = 0.f;
#pragma unroll
    for (int i = 0; i < 8; ++i) { s += ss[i]; sq += ssq[i]; }
    const float mu   = s * (1.f / D_MODEL);
    const float var  = sq * (1.f / D_MODEL) - mu * mu;
    const float rstd = rsqrtf(var + 1e-5f);
    unsigned short* orow = xn + (size_t)row * D_MODEL;
#pragma unroll
    for (int i = 0; i < 4; ++i) {
        int c = threadIdx.x + i * 256;
        orow[c] = f2bf((v[i] - mu) * rstd * gamma[c] + beta[c]);
    }
}

// ---------------------------------------------------------------- bf16 WMMA GEMM: C[m,n] = sum_k A[m,k]*W[n,k]
// Wave tile: 32(M) x 64(N); software-pipelined K loop (step 32).
// mode 0: plain   mode 1: softplus(acc + bias[n])   mode 2: acc + resid[m*ldc+n]
__global__ __launch_bounds__(256) void k_gemm_wmma(const unsigned short* __restrict__ A, int lda,
                                                   const unsigned short* __restrict__ W, int ldw,
                                                   float* __restrict__ C, int ldc,
                                                   int M, int N, int K, int mode,
                                                   const float* __restrict__ bias,
                                                   const float* __restrict__ resid) {
    const int lane   = threadIdx.x & 31;
    const int ntiles = (N + 63) >> 6;     // 64-wide N tiles
    const int mtiles = M >> 5;            // 32-row  M tiles
    const int gw = blockIdx.x * 8 + (threadIdx.x >> 5);   // wave-uniform
    if (gw >= mtiles * ntiles) return;
    const int mt = gw / ntiles;
    const int nt = gw - mt * ntiles;

    const int lh = lane & 15;             // row/col selector within tile
    const int lg = lane >> 4;             // half-wave selector
    const unsigned short* arow0 = A + (size_t)(mt * 32 + lh) * lda;
    const unsigned short* arow1 = arow0 + (size_t)16 * lda;
    const int koA = lg * 8;               // A frag: K = koA..koA+7, koA+16..koA+23
    const int koB = lg * 8;               // B frag: K = koB..koB+7, koB+16..koB+23

    // Clamp OOB weight rows to a valid row; garbage accumulates in acc but is
    // never stored (stores are guarded). Keeps EXEC full and loops branch-free.
    const unsigned short* wr[4];
#pragma unroll
    for (int j = 0; j < 4; ++j) {
        int n = nt * 64 + j * 16 + lh;
        wr[j] = W + (size_t)(n < N ? n : (N - 1)) * ldw;
    }

    const v8f zacc = {0.f, 0.f, 0.f, 0.f, 0.f, 0.f, 0.f, 0.f};
    v8f acc[2][4] = {{zacc, zacc, zacc, zacc}, {zacc, zacc, zacc, zacc}};

    auto loadA = [&](FragBF& f, const unsigned short* r, int k) {
        f.q[0] = *(const v4u*)(r + k + koA);
        f.q[1] = *(const v4u*)(r + k + koA + 16);
    };
    auto loadB = [&](FragBF& f, const unsigned short* r, int k) {
        f.q[0] = *(const v4u*)(r + k + koB);
        f.q[1] = *(const v4u*)(r + k + koB + 16);
    };

    FragBF a0, a1, b0, b1, b2, b3;
    loadA(a0, arow0, 0); loadA(a1, arow1, 0);
    loadB(b0, wr[0], 0); loadB(b1, wr[1], 0);
    loadB(b2, wr[2], 0); loadB(b3, wr[3], 0);

    for (int k = 32; k < K; k += 32) {
        FragBF na0, na1, nb0, nb1, nb2, nb3;
        loadA(na0, arow0, k); loadA(na1, arow1, k);     // prefetch next K-step
        loadB(nb0, wr[0], k); loadB(nb1, wr[1], k);
        loadB(nb2, wr[2], k); loadB(nb3, wr[3], k);

        acc[0][0] = __builtin_amdgcn_wmma_f32_16x16x32_bf16(false, a0.bf, false, b0.bf, (short)0, acc[0][0], false, false);
        acc[0][1] = __builtin_amdgcn_wmma_f32_16x16x32_bf16(false, a0.bf, false, b1.bf, (short)0, acc[0][1], false, false);
        acc[0][2] = __builtin_amdgcn_wmma_f32_16x16x32_bf16(false, a0.bf, false, b2.bf, (short)0, acc[0][2], false, false);
        acc[0][3] = __builtin_amdgcn_wmma_f32_16x16x32_bf16(false, a0.bf, false, b3.bf, (short)0, acc[0][3], false, false);
        acc[1][0] = __builtin_amdgcn_wmma_f32_16x16x32_bf16(false, a1.bf, false, b0.bf, (short)0, acc[1][0], false, false);
        acc[1][1] = __builtin_amdgcn_wmma_f32_16x16x32_bf16(false, a1.bf, false, b1.bf, (short)0, acc[1][1], false, false);
        acc[1][2] = __builtin_amdgcn_wmma_f32_16x16x32_bf16(false, a1.bf, false, b2.bf, (short)0, acc[1][2], false, false);
        acc[1][3] = __builtin_amdgcn_wmma_f32_16x16x32_bf16(false, a1.bf, false, b3.bf, (short)0, acc[1][3], false, false);

        a0 = na0; a1 = na1; b0 = nb0; b1 = nb1; b2 = nb2; b3 = nb3;
    }
    // final K-step
    acc[0][0] = __builtin_amdgcn_wmma_f32_16x16x32_bf16(false, a0.bf, false, b0.bf, (short)0, acc[0][0], false, false);
    acc[0][1] = __builtin_amdgcn_wmma_f32_16x16x32_bf16(false, a0.bf, false, b1.bf, (short)0, acc[0][1], false, false);
    acc[0][2] = __builtin_amdgcn_wmma_f32_16x16x32_bf16(false, a0.bf, false, b2.bf, (short)0, acc[0][2], false, false);
    acc[0][3] = __builtin_amdgcn_wmma_f32_16x16x32_bf16(false, a0.bf, false, b3.bf, (short)0, acc[0][3], false, false);
    acc[1][0] = __builtin_amdgcn_wmma_f32_16x16x32_bf16(false, a1.bf, false, b0.bf, (short)0, acc[1][0], false, false);
    acc[1][1] = __builtin_amdgcn_wmma_f32_16x16x32_bf16(false, a1.bf, false, b1.bf, (short)0, acc[1][1], false, false);
    acc[1][2] = __builtin_amdgcn_wmma_f32_16x16x32_bf16(false, a1.bf, false, b2.bf, (short)0, acc[1][2], false, false);
    acc[1][3] = __builtin_amdgcn_wmma_f32_16x16x32_bf16(false, a1.bf, false, b3.bf, (short)0, acc[1][3], false, false);

#pragma unroll
    for (int i = 0; i < 2; ++i) {
        const int m0 = mt * 32 + i * 16 + lg * 8;   // this lane: rows m0..m0+7, one column
#pragma unroll
        for (int j = 0; j < 4; ++j) {
            int n = nt * 64 + j * 16 + lh;
            if (n >= N) continue;
#pragma unroll
            for (int r = 0; r < 8; ++r) {
                float v = acc[i][j][r];
                if (mode == 1) {
                    float t = v + bias[n];
                    v = (t > 20.f) ? t : log1pf(__expf(t));
                } else if (mode == 2) {
                    v += resid[(size_t)(m0 + r) * ldc + n];
                }
                C[(size_t)(m0 + r) * ldc + n] = v;
            }
        }
    }
}

// ---------------------------------------------------------------- causal depthwise conv (K=4) + SiLU
__global__ __launch_bounds__(256) void k_conv_silu(const float* __restrict__ xz,  // [NTOK, 4096], x_in = cols 0..2047
                                                   const float* __restrict__ cw,  // [2048, 4]
                                                   const float* __restrict__ cb,  // [2048]
                                                   float* __restrict__ u,
                                                   unsigned short* __restrict__ ubf) {
    int i = blockIdx.x * 256 + threadIdx.x;          // over NTOK * D_INNER
    int d = i & (D_INNER - 1);
    int t = i >> 11;                                  // token = b*L + l
    int l = t & (L_SEQ - 1);
    float acc = cb[d];
#pragma unroll
    for (int j = 0; j < 4; ++j) {
        int ll = l - 3 + j;
        if (ll >= 0) acc += cw[d * 4 + j] * xz[(size_t)(t - 3 + j) * 4096 + d];
    }
    float s = acc / (1.f + __expf(-acc));             // silu
    u[i]   = s;
    ubf[i] = f2bf(s);
}

// ---------------------------------------------------------------- selective scan, fused +u*D and silu(z) gate
__global__ __launch_bounds__(256) void k_scan(const float* __restrict__ dt,     // [NTOK, 2048]
                                              const float* __restrict__ u,      // [NTOK, 2048]
                                              const float* __restrict__ xdbl,   // [NTOK, 96]: B=64..79, C=80..95
                                              const float* __restrict__ xz,     // [NTOK, 4096]: z = cols 2048..4095
                                              const float* __restrict__ A_log,  // [2048, 16]
                                              const float* __restrict__ Dp,     // [2048]
                                              unsigned short* __restrict__ ybf) {
    const int d = (blockIdx.x & 7) * 256 + threadIdx.x;
    const int b = blockIdx.x >> 3;
    float Arow[D_STATE], h[D_STATE];
#pragma unroll
    for (int n = 0; n < D_STATE; ++n) { Arow[n] = -__expf(A_log[d * D_STATE + n]); h[n] = 0.f; }
    const float Dd = Dp[d];
    const size_t tbase = (size_t)b * L_SEQ;
    for (int l = 0; l < L_SEQ; ++l) {
        const size_t t = tbase + l;
        const float dtv = dt[t * D_INNER + d];
        const float uv  = u[t * D_INNER + d];
        const float dtu = dtv * uv;
        const float* bc = xdbl + t * 96 + DT_RANK;    // wave-uniform addresses -> broadcast loads
        float y = 0.f;
#pragma unroll
        for (int n = 0; n < D_STATE; ++n) {
            float dA = __expf(dtv * Arow[n]);
            h[n] = h[n] * dA + dtu * bc[n];
            y += h[n] * bc[D_STATE + n];
        }
        y += uv * Dd;
        float zv = xz[t * 4096 + D_INNER + d];
        y *= zv / (1.f + __expf(-zv));                // gate with silu(z)
        ybf[t * D_INNER + d] = f2bf(y);
    }
}

// ----------------------------------------------------------------
static inline size_t align256(size_t x) { return (x + 255) & ~(size_t)255; }

static void launch_gemm(const unsigned short* A, int lda, const unsigned short* W, int ldw,
                        float* C, int ldc, int M, int N, int K, int mode,
                        const float* bias, const float* resid, hipStream_t stream) {
    long long waves = (long long)((N + 63) / 64) * (M / 32);
    int blocks = (int)((waves + 7) / 8);
    k_gemm_wmma<<<blocks, 256, 0, stream>>>(A, lda, W, ldw, C, ldc, M, N, K, mode, bias, resid);
}

extern "C" void kernel_launch(void* const* d_in, const int* in_sizes, int n_in,
                              void* d_out, int out_size, void* d_ws, size_t ws_size,
                              hipStream_t stream) {
    (void)in_sizes; (void)n_in; (void)out_size; (void)ws_size;
    const float* x      = (const float*)d_in[0];
    const float* ln_g   = (const float*)d_in[1];
    const float* ln_b   = (const float*)d_in[2];
    const float* w_in   = (const float*)d_in[3];   // [4096, 1024]
    const float* conv_w = (const float*)d_in[4];   // [2048, 4]
    const float* conv_b = (const float*)d_in[5];   // [2048]
    const float* w_xp   = (const float*)d_in[6];   // [96, 2048]
    const float* w_dt   = (const float*)d_in[7];   // [2048, 64]
    const float* dt_b   = (const float*)d_in[8];   // [2048]
    const float* A_log  = (const float*)d_in[9];   // [2048, 16]
    const float* Dp     = (const float*)d_in[10];  // [2048]
    const float* w_out  = (const float*)d_in[11];  // [1024, 2048]
    float* out = (float*)d_out;                    // [NTOK, 1024]

    char* ws = (char*)d_ws;
    size_t off = 0;
    auto carve = [&](size_t bytes) { char* p = ws + off; off = align256(off + bytes); return p; };
    unsigned short* xn_bf   = (unsigned short*)carve((size_t)NTOK * D_MODEL * 2);
    float*          xz      = (float*)         carve((size_t)NTOK * 2 * D_INNER * 4);
    float*          u       = (float*)         carve((size_t)NTOK * D_INNER * 4);
    unsigned short* u_bf    = (unsigned short*)carve((size_t)NTOK * D_INNER * 2);
    float*          xdbl    = (float*)         carve((size_t)NTOK * 96 * 4);
    unsigned short* dtr_bf  = (unsigned short*)carve((size_t)NTOK * DT_RANK * 2);
    float*          dt      = (float*)         carve((size_t)NTOK * D_INNER * 4);
    unsigned short* y_bf    = (unsigned short*)carve((size_t)NTOK * D_INNER * 2);
    unsigned short* win_bf  = (unsigned short*)carve((size_t)2 * D_INNER * D_MODEL * 2);
    unsigned short* wxp_bf  = (unsigned short*)carve((size_t)96 * D_INNER * 2);
    unsigned short* wdt_bf  = (unsigned short*)carve((size_t)D_INNER * DT_RANK * 2);
    unsigned short* wout_bf = (unsigned short*)carve((size_t)D_MODEL * D_INNER * 2);

    // weight converts (f32 -> bf16)
    { int n = 2 * D_INNER * D_MODEL; k_f32_to_bf16<<<(n + 255) / 256, 256, 0, stream>>>(w_in,  win_bf,  n); }
    { int n = 96 * D_INNER;          k_f32_to_bf16<<<(n + 255) / 256, 256, 0, stream>>>(w_xp,  wxp_bf,  n); }
    { int n = D_INNER * DT_RANK;     k_f32_to_bf16<<<(n + 255) / 256, 256, 0, stream>>>(w_dt,  wdt_bf,  n); }
    { int n = D_MODEL * D_INNER;     k_f32_to_bf16<<<(n + 255) / 256, 256, 0, stream>>>(w_out, wout_bf, n); }

    // 1) layernorm -> xn (bf16)
    k_layernorm<<<NTOK, 256, 0, stream>>>(x, ln_g, ln_b, xn_bf);

    // 2) in_proj: xz = xn @ W_in^T   [NTOK, 4096]
    launch_gemm(xn_bf, D_MODEL, win_bf, D_MODEL, xz, 2 * D_INNER,
                NTOK, 2 * D_INNER, D_MODEL, 0, nullptr, nullptr, stream);

    // 3) causal conv + silu -> u (f32 + bf16)
    { int n = NTOK * D_INNER; k_conv_silu<<<n / 256, 256, 0, stream>>>(xz, conv_w, conv_b, u, u_bf); }

    // 4) x_proj: x_dbl = u @ W_xp^T  [NTOK, 96]
    launch_gemm(u_bf, D_INNER, wxp_bf, D_INNER, xdbl, 96,
                NTOK, 96, D_INNER, 0, nullptr, nullptr, stream);

    // 5) dt_raw (cols 0..63) -> bf16, then dt = softplus(dt_raw @ W_dt^T + b)  [NTOK, 2048]
    { int n = NTOK * DT_RANK; k_slice_bf16<<<(n + 255) / 256, 256, 0, stream>>>(xdbl, 96, 0, dtr_bf, DT_RANK, NTOK); }
    launch_gemm(dtr_bf, DT_RANK, wdt_bf, DT_RANK, dt, D_INNER,
                NTOK, D_INNER, DT_RANK, 1, dt_b, nullptr, stream);

    // 6) selective scan + u*D + silu(z) gate -> y (bf16)
    k_scan<<<B_SZ * (D_INNER / 256), 256, 0, stream>>>(dt, u, xdbl, xz, A_log, Dp, y_bf);

    // 7) out_proj + residual: out = x + y @ W_out^T  [NTOK, 1024]
    launch_gemm(y_bf, D_INNER, wout_bf, D_INNER, out, D_MODEL,
                NTOK, D_MODEL, D_INNER, 2, nullptr, x, stream);
}
